// InfiniteMemory_49374944035487
// MI455X (gfx1250) — matrix operand order repaired and checked by reading, not compiled, not run on hardware
//
#include <hip/hip_runtime.h>
#include <hip/hip_bf16.h>
#include <math.h>

typedef __bf16 bf16_t;
typedef __attribute__((ext_vector_type(16))) bf16_t v16bf;
typedef __attribute__((ext_vector_type(8)))  bf16_t v8bf;
typedef __attribute__((ext_vector_type(8)))  float  v8f;

#define Bq   64
#define Hh   2048
#define Nmem 131072
#define Ktop 100

__device__ __forceinline__ float gelu_exact(float x) {
    return 0.5f * x * (1.0f + erff(x * 0.70710678118654752f));
}

// Load one 16(M)x32(K) bf16 A fragment from row-major A (leading dim = lda elems).
// Layout: lanes 0-15 -> M=lane, K=0..15 ; lanes 16-31 -> K=16..31 (contiguous, matches B).
__device__ __forceinline__ v16bf load_a_frag(const bf16_t* __restrict__ A,
                                             int m0, int k0, int lda, int lane) {
    int m  = m0 + (lane & 15);
    int hi = lane >> 4;
    const bf16_t* p = A + (size_t)m * lda + k0 + hi * 16;
    v8bf lo = *(const v8bf*)(p);
    v8bf hh = *(const v8bf*)(p + 8);
    v16bf a;
#pragma unroll
    for (int e = 0; e < 8; ++e) { a[e] = lo[e]; a[e + 8] = hh[e]; }
    return a;
}

// C[64,Nw] = act(A_bf16[64,Kdim] @ W_f32[Kdim,Nw] + bias). One wave per 16-col tile,
// computing the full M=64 slab (4 WMMA accumulators), bf16 inputs / f32 accumulate.
__global__ void __launch_bounds__(32)
wmma_mlp_gemm(const bf16_t* __restrict__ A, const float* __restrict__ W,
              const float* __restrict__ bias, int Kdim, int Nw,
              float* __restrict__ outF, bf16_t* __restrict__ outB, int act)
{
    int lane = threadIdx.x & 31;
    int n0   = blockIdx.x * 16;
    int hi   = lane >> 4;
    int ncol = n0 + (lane & 15);

    v8f acc[4] = {};
    for (int k0 = 0; k0 < Kdim; k0 += 32) {
        // B fragment: lane holds column ncol, K = k0 + hi*16 + e. For fixed e the
        // 16 lanes of a half-wave read 64B contiguous (coalesced weight stream).
        int kb = k0 + hi * 16;
        v16bf bf;
#pragma unroll
        for (int e = 0; e < 16; ++e)
            bf[e] = (bf16_t)W[(size_t)(kb + e) * Nw + ncol];
#pragma unroll
        for (int m = 0; m < 4; ++m) {
            v16bf af = load_a_frag(A, m * 16, k0, Kdim, lane);
            acc[m] = __builtin_amdgcn_wmma_f32_16x16x32_bf16(
                false, af, false, bf, (short)0, acc[m], false, false);
        }
    }

    float bv = bias[ncol];
#pragma unroll
    for (int m = 0; m < 4; ++m) {
#pragma unroll
        for (int r = 0; r < 8; ++r) {
            int row = m * 16 + hi * 8 + r;      // D layout: VGPR r -> M = hi*8 + r
            float v = acc[m][r] + bv;
            if (act == 1) v = gelu_exact(v);
            if (outF) outF[(size_t)row * Nw + ncol] = v;
            if (outB) outB[(size_t)row * Nw + ncol] = (bf16_t)v;
        }
    }
}

// sims[64,Nmem] = encoded_bf16[64,2048] @ memory[Nmem,2048]^T.
// Per block: 64 memory rows. Each K-step's 64x32-f32 tile (8KB) is staged into LDS
// with GLOBAL_LOAD_ASYNC_TO_LDS_B128 (ASYNCcnt-tracked, double-buffered), so the
// 1.07 GB HBM stream runs decoupled from the WMMA issue stream. Async loads of a
// wave complete in order, so s_wait_asynccnt 4 after issuing tile k+1 guarantees
// tile k is resident; the workgroup barrier covers the other waves' portions.
__global__ void __launch_bounds__(128)
sims_wmma(const bf16_t* __restrict__ A, const float* __restrict__ Mem,
          float* __restrict__ S)
{
    __shared__ float tile[2][64 * 32];            // 16 KB double buffer

    const int tid     = threadIdx.x;              // 0..127
    const int lane    = tid & 31;
    const int wave    = tid >> 5;                 // 0..3 -> 16-row n-tile
    const int hi      = lane >> 4;
    const int rowBase = blockIdx.x * 64;
    const unsigned ldsBase = (unsigned)(uintptr_t)(&tile[0][0]);  // AS3 offset

    // Issue one 8KB tile (k-step k0) into buffer `buf`: 4 async b128 per wave.
    auto issue_tile = [&](int buf, int k0) {
#pragma unroll
        for (int j = 0; j < 4; ++j) {
            int chunk = j * 128 + tid;            // 16B chunk id, 512 chunks/tile
            int row   = chunk >> 3;               // 8 chunks per 32-float row
            int col   = (chunk & 7) * 4;
            const float* gp = Mem + (size_t)(rowBase + row) * Hh + k0 + col;
            unsigned ldsa = ldsBase + (unsigned)((buf * 2048 + chunk * 4) * 4);
            unsigned long long ga = (unsigned long long)(uintptr_t)gp;
            asm volatile("global_load_async_to_lds_b128 %0, %1, off"
                         :: "v"(ldsa), "v"(ga) : "memory");
        }
    };

    v8f acc[4] = {};
    issue_tile(0, 0);
    for (int kt = 0; kt < Hh / 32; ++kt) {
        const int cur = kt & 1;
        const int k0  = kt * 32;
        if (kt + 1 < Hh / 32) {
            issue_tile(cur ^ 1, k0 + 32);
            asm volatile("s_wait_asynccnt 0x4" ::: "memory");
        } else {
            asm volatile("s_wait_asynccnt 0x0" ::: "memory");
        }
        __syncthreads();                          // all waves' tile parts resident

        // B fragment from LDS: lane -> local row wave*16 + (lane&15), K = hi*16+e
        const float* lp = &tile[cur][(wave * 16 + (lane & 15)) * 32 + hi * 16];
        float4 f0 = *(const float4*)(lp + 0);
        float4 f1 = *(const float4*)(lp + 4);
        float4 f2 = *(const float4*)(lp + 8);
        float4 f3 = *(const float4*)(lp + 12);
        v16bf bf;
        bf[0]=(bf16_t)f0.x;  bf[1]=(bf16_t)f0.y;  bf[2]=(bf16_t)f0.z;  bf[3]=(bf16_t)f0.w;
        bf[4]=(bf16_t)f1.x;  bf[5]=(bf16_t)f1.y;  bf[6]=(bf16_t)f1.z;  bf[7]=(bf16_t)f1.w;
        bf[8]=(bf16_t)f2.x;  bf[9]=(bf16_t)f2.y;  bf[10]=(bf16_t)f2.z; bf[11]=(bf16_t)f2.w;
        bf[12]=(bf16_t)f3.x; bf[13]=(bf16_t)f3.y; bf[14]=(bf16_t)f3.z; bf[15]=(bf16_t)f3.w;
#pragma unroll
        for (int m = 0; m < 4; ++m) {
            v16bf af = load_a_frag(A, m * 16, k0, Hh, lane);
            acc[m] = __builtin_amdgcn_wmma_f32_16x16x32_bf16(
                false, af, false, bf, (short)0, acc[m], false, false);
        }
        __syncthreads();                          // protect buffer before reuse
    }

    const int nrow = rowBase + wave * 16 + (lane & 15);
#pragma unroll
    for (int m = 0; m < 4; ++m)
#pragma unroll
        for (int r = 0; r < 8; ++r)
            S[(size_t)(m * 16 + hi * 8 + r) * Nmem + nrow] = acc[m][r];
}

// importance = sigmoid(h2[64,1024] @ imp_w2[1024] + b2)
__global__ void __launch_bounds__(256)
imp_final(const float* __restrict__ h2, const float* __restrict__ w2,
          const float* __restrict__ b2, float* __restrict__ out)
{
    int b = blockIdx.x, tid = threadIdx.x;
    float s = 0.f;
    for (int j = tid; j < Hh / 2; j += 256) s += h2[b * (Hh / 2) + j] * w2[j];
    __shared__ float red[256];
    red[tid] = s; __syncthreads();
    for (int st = 128; st > 0; st >>= 1) {
        if (tid < st) red[tid] += red[tid + st];
        __syncthreads();
    }
    if (tid == 0) out[b] = 1.0f / (1.0f + expf(-(red[0] + b2[0])));
}

// Top-100 per row of sims[64,Nmem]. Each of 1024 threads owns 128 elements and a
// cached local argmax; per extraction only the winner's owner rescans 128 elems.
// Tie-break on lowest index (stable, matches lax.top_k); output order descending.
__global__ void __launch_bounds__(1024)
topk_kernel(float* __restrict__ S, float* __restrict__ outV, float* __restrict__ outI)
{
    int b = blockIdx.x, tid = threadIdx.x;
    float* row = S + (size_t)b * Nmem;
    const int per = Nmem / 1024;                 // 128
    int base = tid * per;

    float lv = -__builtin_inff(); int li = base;
    for (int i = 0; i < per; ++i) {
        float v = row[base + i];
        if (v > lv) { lv = v; li = base + i; }
    }

    __shared__ float sv[1024];
    __shared__ int   si[1024];
    for (int t = 0; t < Ktop; ++t) {
        sv[tid] = lv; si[tid] = li;
        __syncthreads();
        for (int s = 512; s > 0; s >>= 1) {
            if (tid < s) {
                float ov = sv[tid + s]; int oi = si[tid + s];
                if (ov > sv[tid] || (ov == sv[tid] && oi < si[tid])) {
                    sv[tid] = ov; si[tid] = oi;
                }
            }
            __syncthreads();
        }
        float wv = sv[0]; int wi = si[0];
        if (tid == 0) {
            outV[b * Ktop + t] = wv;
            outI[b * Ktop + t] = (float)wi;      // indices exact in f32 (< 2^24)
        }
        __syncthreads();                          // all read sv[0]/si[0] before reuse
        if (wi >= base && wi < base + per) {
            row[wi] = -__builtin_inff();
            lv = -__builtin_inff(); li = base;
            for (int i = 0; i < per; ++i) {
                float v = row[base + i];
                if (v > lv) { lv = v; li = base + i; }
            }
        }
    }
}

__global__ void f2bf_kernel(const float* __restrict__ x, bf16_t* __restrict__ y, int n)
{
    int i = blockIdx.x * blockDim.x + threadIdx.x;
    if (i < n) y[i] = (bf16_t)x[i];
}

extern "C" void kernel_launch(void* const* d_in, const int* in_sizes, int n_in,
                              void* d_out, int out_size, void* d_ws, size_t ws_size,
                              hipStream_t stream)
{
    const float* embedding = (const float*)d_in[0];
    const float* memory    = (const float*)d_in[1];
    const float* enc_w1    = (const float*)d_in[2];
    const float* enc_b1    = (const float*)d_in[3];
    const float* enc_w2    = (const float*)d_in[4];
    const float* enc_b2    = (const float*)d_in[5];
    const float* imp_w1    = (const float*)d_in[6];
    const float* imp_b1    = (const float*)d_in[7];
    const float* imp_w2    = (const float*)d_in[8];
    const float* imp_b2    = (const float*)d_in[9];
    float* out = (float*)d_out;

    // Workspace layout (bytes)
    char* ws = (char*)d_ws;
    bf16_t* embB = (bf16_t*)(ws);                            // 64*2048*2  = 256 KB
    bf16_t* h1B  = (bf16_t*)(ws + 262144);                   // 64*4096*2  = 512 KB
    bf16_t* encB = (bf16_t*)(ws + 262144 + 524288);          // 64*2048*2  = 256 KB
    float*  h2   = (float*) (ws + 1048576);                  // 64*1024*4  = 256 KB
    float*  sims = (float*) (ws + 1310720);                  // 64*131072*4 = 32 MB

    // Output layout: encoded[64*2048] | importance[64] | topk_vals[64*100] | topk_idx[64*100]
    float* out_enc = out;
    float* out_imp = out + Bq * Hh;              // 131072
    float* out_val = out_imp + Bq;               // 131136
    float* out_idx = out_val + Bq * Ktop;        // 137536

    // 1) embedding -> bf16
    f2bf_kernel<<<(Bq * Hh + 255) / 256, 256, 0, stream>>>(embedding, embB, Bq * Hh);
    // 2) h1 = GELU(emb @ enc_w1 + b1)  [64,4096] bf16
    wmma_mlp_gemm<<<(2 * Hh) / 16, 32, 0, stream>>>(embB, enc_w1, enc_b1, Hh, 2 * Hh,
                                                    nullptr, h1B, 1);
    // 3) encoded = h1 @ enc_w2 + b2   -> f32 output + bf16 copy for the big GEMM
    wmma_mlp_gemm<<<Hh / 16, 32, 0, stream>>>(h1B, enc_w2, enc_b2, 2 * Hh, Hh,
                                              out_enc, encB, 0);
    // 4) h2 = GELU(encoded @ imp_w1 + b1)  [64,1024] f32
    wmma_mlp_gemm<<<(Hh / 2) / 16, 32, 0, stream>>>(encB, imp_w1, imp_b1, Hh, Hh / 2,
                                                    h2, nullptr, 1);
    // 5) importance = sigmoid(h2 @ imp_w2 + b2)
    imp_final<<<Bq, 256, 0, stream>>>(h2, imp_w2, imp_b2, out_imp);
    // 6) sims = encoded @ memory^T  (async-LDS staged, bandwidth-bound bf16 WMMA)
    sims_wmma<<<Nmem / 64, 128, 0, stream>>>(encB, memory, sims);
    // 7) per-row top-100 (values + indices-as-float)
    topk_kernel<<<Bq, 1024, 0, stream>>>(sims, out_val, out_idx);
}
